// CliffordSharedSimplicialMPNN_md17_69853348102232
// MI455X (gfx1250) — matrix operand order repaired and checked by reading, not compiled, not run on hardware
//
#include <hip/hip_runtime.h>
#include <cmath>

// ---------------------------------------------------------------------------
// CDNA5 (gfx1250) CliffordSharedSimplicialMPNN forward.
// Multivector tensors are PLANAR: [8 components][rows][channels]; for fixed
// component k each mvlinear is a dense row-major GEMM:
//   Y_k[r,o] = sum_i X_k[r,i] * W[o,i,GRADE[k]]  -> v_wmma_f32_16x16x32_f16.
// All GEMM inputs are zero-padded to 32-channel chunks, so every A fragment
// is four aligned float4 loads (no guarded tail path). Grade-0-only inputs
// (node/edge attributes) are stored as a single plane and their chunk is
// skipped uniformly for components k>0 (A is identically zero there).
// Chunk count / weight layout are template parameters; output stores use one
// base pointer with incremental row stride (no per-element 64-bit muls).
// ---------------------------------------------------------------------------

typedef __attribute__((ext_vector_type(16))) _Float16 v16h;
typedef __attribute__((ext_vector_type(8)))  float    v8f;

static const int Bn   = 300;
static const int N0G  = 20;
static const int NN   = 24000;
static const int N0   = 6000, N1 = 12000, N2 = 6000;
static const int EE   = 96000;
static const int NFt  = 10;
static const int HIDC = 32;
static const int FLD  = 96;     // padded leading dim of simplex-feature arena

__constant__ int GRADE_C[8] = {0,1,1,1,2,2,2,3};
__constant__ int MASK_C[8]  = {0,1,2,4,3,5,6,7};
__constant__ int IDX_C[8]   = {0,1,2,4,3,5,6,7};   // mask value -> component index
__constant__ int PERM1_C[2][2] = {{0,1},{1,0}};
__constant__ int PERM2_C[6][3] = {{0,1,2},{0,2,1},{1,0,2},{1,2,0},{2,0,1},{2,1,0}};

// One 32-channel K-chunk of the GEMM input (chunk c covers global K c*32..).
struct GemmSeg {
    const float* ptr;     // base (component-plane 0)
    const int*   idx;     // optional row gather, may be null
    int          ldr;     // row stride (floats)
    int          coff;    // channel offset
    int          pstride; // component-plane stride (0 for single-plane tensors)
    int          k0only;  // 1: tensor is zero for components k>0 -> skip chunk
};

__device__ __forceinline__ float sigm(float v) { return 1.0f / (1.0f + __expf(-v)); }

__device__ __forceinline__ float csign(int a, int b) {
    int s = 0; a >>= 1;
    while (a) { s += __popc(a & b); a >>= 1; }
    return (s & 1) ? -1.0f : 1.0f;
}

__device__ __forceinline__ v16h hzero() {
    v16h z;
#pragma unroll
    for (int i = 0; i < 16; ++i) z[i] = (_Float16)0.0f;
    return z;
}

// A fragment: lane half h needs local K runs [8h,8h+8) and [16+8h,16+8h+8)
// -> four aligned float4 loads, packed f32->f16 converts, one WMMA.
__device__ __forceinline__ v8f wmma_chunk(const float* base, int half,
                                          v16h bfrag, v8f acc)
{
    const float4 q0 = *(const float4*)(base + 8 * half);
    const float4 q1 = *(const float4*)(base + 8 * half + 4);
    const float4 q2 = *(const float4*)(base + 16 + 8 * half);
    const float4 q3 = *(const float4*)(base + 16 + 8 * half + 4);
    v16h af;
    af[0]=(_Float16)q0.x;  af[1]=(_Float16)q0.y;  af[2]=(_Float16)q0.z;  af[3]=(_Float16)q0.w;
    af[4]=(_Float16)q1.x;  af[5]=(_Float16)q1.y;  af[6]=(_Float16)q1.z;  af[7]=(_Float16)q1.w;
    af[8]=(_Float16)q2.x;  af[9]=(_Float16)q2.y;  af[10]=(_Float16)q2.z; af[11]=(_Float16)q2.w;
    af[12]=(_Float16)q3.x; af[13]=(_Float16)q3.y; af[14]=(_Float16)q3.z; af[15]=(_Float16)q3.w;
    return __builtin_amdgcn_wmma_f32_16x16x32_f16(false, af, false, bfrag,
                                                  (short)0, acc, false, false);
}

__device__ __forceinline__ const float* seg_base(const GemmSeg& s, int k, int rc)
{
    const int rr = s.idx ? s.idx[rc] : rc;
    return s.ptr + (size_t)k * s.pstride + (size_t)rr * s.ldr + s.coff;
}

// B layout (32x16): lane half h holds K [8h,8h+8) then [16+8h,16+8h+8),
// matching the A fragment; halves of VGPR v hold consecutive K.
template<int WSUB>
__device__ __forceinline__ v16h load_bfrag(const float* __restrict__ w,
                                           int fi, int fo, int o, int g,
                                           int c, int half)
{
    v16h bf;
#pragma unroll
    for (int v = 0; v < 8; ++v) {
        const int K0 = c * 32 + half * 8 + 2 * v + (v >= 4 ? 8 : 0);
        float w0 = 0.0f, w1 = 0.0f;
        if (o < fo && K0 < fi)
            w0 = WSUB ? w[(o * fi + K0) * 4 + g] : w[o * fi + K0];
        if (o < fo && K0 + 1 < fi)
            w1 = WSUB ? w[(o * fi + K0 + 1) * 4 + g] : w[o * fi + K0 + 1];
        bf[2 * v]     = (_Float16)w0;
        bf[2 * v + 1] = (_Float16)w1;
    }
    return bf;
}

template<int NCH, int WSUB>
__global__ __launch_bounds__(32)
void k_mv_gemm_wmma(GemmSeg s0, GemmSeg s1, GemmSeg s2,
                    const float* __restrict__ w, const float* __restrict__ bias,
                    float* __restrict__ y,
                    int rows, int fi, int fo, int rpw)
{
    const int lane = threadIdx.x & 31;
    const int half = lane >> 4;
    const int nl   = lane & 15;
    const int ntiles = (fo + 15) >> 4;
    int bid = blockIdx.x;
    const int k  = bid & 7; bid >>= 3;
    const int nt = bid % ntiles;
    const int rg = bid / ntiles;
    const int g  = GRADE_C[k];
    const int o  = nt * 16 + nl;

    const bool a0 = !(s0.k0only && k != 0);
    const bool a1 = (NCH >= 2) && !(s1.k0only && k != 0);
    const bool a2 = (NCH >= 3) && !(s2.k0only && k != 0);

    v16h bf0 = hzero(), bf1 = hzero(), bf2 = hzero();
    if (a0) bf0 = load_bfrag<WSUB>(w, fi, fo, o, g, 0, half);
    if (NCH >= 2 && a1) bf1 = load_bfrag<WSUB>(w, fi, fo, o, g, 1, half);
    if (NCH >= 3 && a2) bf2 = load_bfrag<WSUB>(w, fi, fo, o, g, 2, half);

    float bia = 0.0f;
    if (bias != nullptr && k == 0 && o < fo) bia = bias[o];

    const bool fullN = (nt * 16 + 16) <= fo;
    float* const yk = y + (size_t)k * ((size_t)rows * fo);

    for (int t = 0; t < rpw; ++t) {
        const int row0 = (rg * rpw + t) * 16;
        if (row0 >= rows) break;                       // uniform
        // Out-of-range rows: clamp the load row (A row m only feeds D row m),
        // mask on store.
        const int r  = row0 + nl;
        const int rc = (r < rows) ? r : (rows - 1);

        v8f acc;
#pragma unroll
        for (int v = 0; v < 8; ++v) acc[v] = 0.0f;

        if (a0)             acc = wmma_chunk(seg_base(s0, k, rc), half, bf0, acc);
        if (NCH >= 2 && a1) acc = wmma_chunk(seg_base(s1, k, rc), half, bf1, acc);
        if (NCH >= 3 && a2) acc = wmma_chunk(seg_base(s2, k, rc), half, bf2, acc);

        // D row (row0 + v + 8*half): one base pointer, stride fo per v.
        const int rbase = row0 + 8 * half;
        float* yp = yk + (size_t)rbase * fo + o;
        if (fullN && (row0 + 16 <= rows)) {
#pragma unroll
            for (int v = 0; v < 8; ++v) { *yp = acc[v] + bia; yp += fo; }
        } else {
#pragma unroll
            for (int v = 0; v < 8; ++v) {
                if (rbase + v < rows && o < fo) *yp = acc[v] + bia;
                yp += fo;
            }
        }
    }
}

// -------------------------- elementwise kernels ----------------------------
// Planar addressing: (k, r, o) of [8][rows][fo] at buf[k*PS + r*fo + o].

__global__ void k_zero(float* p, int n) {
    int t = blockIdx.x * blockDim.x + threadIdx.x;
    if (t < n) p[t] = 0.0f;
}

__global__ void k_add(float* x, const float* d, int n) {
    int t = blockIdx.x * blockDim.x + threadIdx.x;
    if (t < n) x[t] += d[t];
}

__global__ void k_mvsilu(float* x, const float* a, const float* b, int total, int fo) {
    int t = blockIdx.x * blockDim.x + threadIdx.x;
    if (t >= total) return;
    const int PS = total;
    int o = t % fo;
    float v[8];
#pragma unroll
    for (int k = 0; k < 8; ++k) v[k] = x[k * PS + t];
    float inv[4] = { v[0],
                     v[1]*v[1] + v[2]*v[2] + v[3]*v[3],
                     v[4]*v[4] + v[5]*v[5] + v[6]*v[6],
                     v[7]*v[7] };
    float gate[4];
#pragma unroll
    for (int g = 0; g < 4; ++g) gate[g] = sigm(a[o * 4 + g] * inv[g] + b[o * 4 + g]);
#pragma unroll
    for (int k = 0; k < 8; ++k) x[k * PS + t] = v[k] * gate[GRADE_C[k]];
}

__global__ void k_norm(float* x, const float* a, int total, int fo) {
    int t = blockIdx.x * blockDim.x + threadIdx.x;
    if (t >= total) return;
    const int PS = total;
    int o = t % fo;
    float v[8];
#pragma unroll
    for (int k = 0; k < 8; ++k) v[k] = x[k * PS + t];
    float q[4] = { v[0]*v[0],
                   v[1]*v[1] + v[2]*v[2] + v[3]*v[3],
                   v[4]*v[4] + v[5]*v[5] + v[6]*v[6],
                   v[7]*v[7] };
    float nn[4];
#pragma unroll
    for (int g = 0; g < 4; ++g) {
        float n = sqrtf(q[g] + 1e-12f);
        nn[g] = sigm(a[o * 4 + g]) * (n - 1.0f) + 1.0f;
    }
#pragma unroll
    for (int k = 0; k < 8; ++k) x[k * PS + t] = v[k] / (nn[GRADE_C[k]] + 1e-6f);
}

// t := (l1 + geometric_product(t, xr)) / sqrt(2)   (in-place on t)
__global__ void k_sgp(float* t, const float* xr, const float* l1,
                      const float* gpw, int total, int fo) {
    int tid = blockIdx.x * blockDim.x + threadIdx.x;
    if (tid >= total) return;
    const int PS = total;
    int o = tid % fo;
    float x[8], r[8], acc[8];
#pragma unroll
    for (int k = 0; k < 8; ++k) {
        x[k] = t[k * PS + tid]; r[k] = xr[k * PS + tid]; acc[k] = l1[k * PS + tid];
    }
    const float* W = gpw + o * 64;
#pragma unroll
    for (int i = 0; i < 8; ++i) {
#pragma unroll
        for (int j = 0; j < 8; ++j) {
            int mi = MASK_C[i], mj = MASK_C[j];
            int kk = IDX_C[mi ^ mj];
            acc[kk] += csign(mi, mj) *
                       W[GRADE_C[i]*16 + GRADE_C[j]*4 + GRADE_C[kk]] * x[i] * r[j];
        }
    }
#pragma unroll
    for (int k = 0; k < 8; ++k) t[k * PS + tid] = acc[k] * 0.70710678118654752f;
}

__global__ void k_seg_add(const float* msg, const int* dst, float* agg, int n) {
    int t = blockIdx.x * blockDim.x + threadIdx.x;   // n = 8*E*32
    if (t >= n) return;
    int o = t & 31;
    int e = (t >> 5) % EE;
    int p = t / (EE * 32);
#if defined(__HIP_DEVICE_COMPILE__)
    unsafeAtomicAdd(&agg[(size_t)p * (NN * 32) + (size_t)dst[e] * 32 + o], msg[t]);
#else
    atomicAdd(&agg[(size_t)p * (NN * 32) + (size_t)dst[e] * 32 + o], msg[t]);
#endif
}

// x[:, idx[i], :] = sum_p T[:, i*f+p, :]
__global__ void k_perm_sum_set(float* x, const float* T, const int* idx, int nd, int f) {
    int t = blockIdx.x * blockDim.x + threadIdx.x;   // over 8*nd*32
    if (t >= 8 * nd * 32) return;
    int o = t & 31;
    int i = (t >> 5) % nd;
    int k = t / (nd * 32);
    const int PST = nd * f * 32;
    float s = 0.0f;
    for (int p = 0; p < f; ++p) s += T[(size_t)k * PST + (size_t)(i * f + p) * 32 + o];
    int n = idx ? idx[i] : i;
    x[(size_t)k * (NN * 32) + (size_t)n * 32 + o] = s;
}

__global__ void k_gmean(const float* loc, const int* idx0, float* gmean) {
    int t = blockIdx.x * blockDim.x + threadIdx.x;
    if (t >= Bn * 3) return;
    int b = t / 3, c = t % 3;
    float s = 0.0f;
    for (int g = 0; g < N0G; ++g) {
        int n = idx0[b * N0G + g];
        for (int f = 0; f < NFt; ++f) s += loc[(n * NFt + f) * 3 + c];
    }
    gmean[t] = s / (float)(N0G * NFt);
}

__global__ void k_pos(const float* loc, const float* gmean, const int* batch, float* pos) {
    int t = blockIdx.x * blockDim.x + threadIdx.x;
    if (t >= NN * NFt * 3) return;
    int n = t / (NFt * 3), c = t % 3;
    pos[t] = loc[t] - gmean[batch[n] * 3 + c];
}

// node_attr: single plane [NN][32] (only component 0 is nonzero)
__global__ void k_node_attr(const float* sim_emb, const int* types, float* na) {
    int t = blockIdx.x * blockDim.x + threadIdx.x;
    if (t >= NN * 32) return;
    int o = t & 31, n = t >> 5;
    na[t] = (o < 3) ? sim_emb[types[n] * 3 + o] : 0.0f;
}

// edge_attr: single plane [E][32] (only component 0 is nonzero)
__global__ void k_edge_attr(const float* sim_emb, const int* types, const int* ei, float* ea) {
    int t = blockIdx.x * blockDim.x + threadIdx.x;
    if (t >= EE * 32) return;
    int o = t & 31, e = t >> 5;
    float v = 0.0f;
    if (o < 6) {
        int node = (o < 3) ? ei[e] : ei[EE + e];
        int cc   = (o < 3) ? o : o - 3;
        v = sim_emb[types[node] * 3 + cc];
    }
    ea[t] = v;
}

__device__ __forceinline__ int perm_at(int d, int p, int s) {
    if (d == 0) return 0;
    if (d == 1) return PERM1_C[p][s];
    return PERM2_C[p][s];
}

// F planar [8][nd*f][FLD]; only channels [0, 3*(d+1)*NF) written, pad stays 0.
__global__ void k_simplex(const float* pos, const float* vel, const float* ch,
                          const int* x_ind, float* F, int nd, int f, int d) {
    const int chn = 3 * (d + 1) * NFt;
    const int rowsT = nd * f;
    long total = (long)rowsT * chn * 8;
    long t = (long)blockIdx.x * blockDim.x + threadIdx.x;
    if (t >= total) return;
    int  i   = (int)(t % chn);
    long t2  = t / chn;
    int  k   = (int)(t2 % 8);
    int  row = (int)(t2 / 8);
    int  p   = row % f;
    int  ndi = row / f;
    int sec = i / ((d + 1) * NFt);
    int li  = i % ((d + 1) * NFt);
    int s   = li / NFt, ff = li % NFt;
    int node = x_ind[ndi * (d + 1) + perm_at(d, p, s)];
    float val = 0.0f;
    if (sec < 2) {
        if (k >= 1 && k <= 3)
            val = (sec == 0 ? pos : vel)[(node * NFt + ff) * 3 + (k - 1)];
    } else {
        if (k == 0) val = ch[node * NFt + ff];
    }
    F[(size_t)k * ((size_t)rowsT * FLD) + (size_t)row * FLD + i] = val;
}

// Y planar [8][N0][10]; pred = components 1..3
__global__ void k_out(const float* loc, const int* idx0, const float* Y, float* out) {
    int t = blockIdx.x * blockDim.x + threadIdx.x;
    if (t >= N0 * NFt * 3) return;
    int r = t / 30, q = t % 30, ff = q / 3, c = q % 3;
    const int PS = N0 * 10;
    out[t] = loc[idx0[r] * 30 + q] + Y[(size_t)(1 + c) * PS + (size_t)r * 10 + ff];
}

// ------------------------------ host side ----------------------------------

static inline int cdiv(long a, long b) { return (int)((a + b - 1) / b); }

struct MvW    { const float* b; const float* w; };
struct BlockP { MvW lin; const float* gp_w; MvW left; const float* right_w;
                const float* norm_a; const float* silu_a; const float* silu_b; };

struct Cur {
    void* const* in; int i;
    const float* f()  { return (const float*)in[i++]; }
    const int*   s()  { return (const int*)in[i++]; }
};

static BlockP readBlock(Cur& c) {
    BlockP b;
    b.lin.b = c.f(); b.lin.w = c.f();          // lin: {b, w}
    b.gp_w  = c.f();                           // sgp.gp_w
    b.left.b = c.f(); b.left.w = c.f();        // sgp.lin_left: {b, w}
    b.right_w = c.f();                         // sgp.lin_right: {w}
    b.norm_a  = c.f();                         // sgp.norm: {a}
    b.silu_a = c.f(); b.silu_b = c.f();        // silu: {a, b}
    return b;
}

static GemmSeg segZ() { return GemmSeg{nullptr, nullptr, 0, 0, 0, 0}; }
static GemmSeg seg32(const float* p, const int* idx, int rowsTotal) {
    return GemmSeg{p, idx, 32, 0, rowsTotal * 32, 0};
}

static void mv_gemm(hipStream_t st, int nch, int wsub,
                    GemmSeg s0, GemmSeg s1, GemmSeg s2,
                    const float* w, const float* b, float* y,
                    int rows, int fi, int fo) {
    const int RPW = 4;
    int ntiles = (fo + 15) / 16;
    int rtiles = (rows + 15) / 16;
    dim3 grid(((rtiles + RPW - 1) / RPW) * ntiles * 8);
#define LNCH(N, S) k_mv_gemm_wmma<N, S><<<grid, 32, 0, st>>>(s0, s1, s2, w, b, y, rows, fi, fo, RPW)
    if (wsub) { if (nch == 1) LNCH(1, 1); else if (nch == 2) LNCH(2, 1); else LNCH(3, 1); }
    else      { if (nch == 1) LNCH(1, 0); else LNCH(2, 0); }
#undef LNCH
}

static void run_block(hipStream_t st, const BlockP& bp, int rows,
                      GemmSeg s0, GemmSeg s1, GemmSeg s2, int nch, int fi,
                      float* OUT, float* W1, float* W2) {
    GemmSeg z = segZ();
    mv_gemm(st, nch, 1, s0, s1, s2, bp.lin.w, bp.lin.b, OUT, rows, fi, HIDC);
    int total = rows * HIDC;
    k_mvsilu<<<cdiv(total, 256), 256, 0, st>>>(OUT, bp.silu_a, bp.silu_b, total, HIDC);
    GemmSeg so = seg32(OUT, nullptr, rows);
    mv_gemm(st, 1, 1, so, z, z, bp.right_w, nullptr, W1, rows, HIDC, HIDC);
    k_norm<<<cdiv(total, 256), 256, 0, st>>>(W1, bp.norm_a, total, HIDC);
    mv_gemm(st, 1, 1, so, z, z, bp.left.w, bp.left.b, W2, rows, HIDC, HIDC);
    k_sgp<<<cdiv(total, 256), 256, 0, st>>>(OUT, W1, W2, bp.gp_w, total, HIDC);
}

extern "C" void kernel_launch(void* const* d_in, const int* in_sizes, int n_in,
                              void* d_out, int out_size, void* d_ws, size_t ws_size,
                              hipStream_t stream) {
    (void)in_sizes; (void)n_in; (void)out_size; (void)ws_size;

    // ---- unpack params in JAX pytree (sorted-key) order ----
    Cur c{d_in, 0};
    MvW cl0;  cl0.b = c.f();  cl0.w = c.f();
    BlockP cl1b = readBlock(c);
    BlockP cl2b[2] = { readBlock(c), readBlock(c) };
    BlockP eg_e[5][2], eg_n[5][2];
    for (int l = 0; l < 5; ++l) {
        eg_e[l][0] = readBlock(c); eg_e[l][1] = readBlock(c);
        eg_n[l][0] = readBlock(c); eg_n[l][1] = readBlock(c);
    }
    MvW feat;  feat.b = c.f();  feat.w = c.f();
    MvW plin;  plin.b = c.f();  plin.w = c.f();
    BlockP pmlp = readBlock(c);
    const float* sim_emb = c.f();

    const float* loc     = c.f();
    const float* vel     = c.f();
    const float* charges = c.f();
    const int* node_types = c.s();
    const int* ei   = c.s();                // [2, E]
    const int* xi0  = c.s();
    const int* xi1  = c.s();
    const int* xi2  = c.s();
    const int* id0  = c.s();
    const int* id1  = c.s();
    const int* id2  = c.s();
    const int* batch = c.s();
    const int* node0_batch = c.s(); (void)node0_batch;

    const int* src = ei;
    const int* dst = ei + EE;

    // ---- workspace layout (~465 MB fp32) ----
    float* W = (float*)d_ws; size_t off = 0;
    auto alloc = [&](size_t n) { float* p = W + off; off += n; return p; };
    float* gmean = alloc(Bn * 3);
    float* pos   = alloc((size_t)NN * NFt * 3);
    float* nattr = alloc((size_t)NN * 32);           // single plane, k==0 only
    float* eattr = alloc((size_t)EE * 32);           // single plane, k==0 only
    float* x     = alloc((size_t)8 * NN * 32);
    float* agg   = alloc((size_t)8 * NN * 32);
    float* Fb    = alloc((size_t)8 * 36000 * FLD);   // simplex feats (ld=96)
    float* Tb    = alloc((size_t)8 * EE * 32);
    float* Rb    = alloc((size_t)8 * EE * 32);
    float* Lb    = alloc((size_t)8 * EE * 32);

    GemmSeg z = segZ();
    GemmSeg na{nattr, nullptr, 32, 0, 0, 1};
    GemmSeg ea{eattr, nullptr, 32, 0, 0, 1};

    // ---- geometry preprocessing ----
    k_gmean<<<cdiv(Bn * 3, 256), 256, 0, stream>>>(loc, id0, gmean);
    k_pos  <<<cdiv((long)NN * NFt * 3, 256), 256, 0, stream>>>(loc, gmean, batch, pos);
    k_node_attr<<<cdiv((long)NN * 32, 256), 256, 0, stream>>>(sim_emb, node_types, nattr);
    k_edge_attr<<<cdiv((long)EE * 32, 256), 256, 0, stream>>>(sim_emb, node_types, ei, eattr);
    k_zero<<<cdiv((long)8 * 36000 * FLD, 256), 256, 0, stream>>>(Fb, 8 * 36000 * FLD);

    // ---- simplex encoders -> x ----
    // d0: 6000 rows, 30 ch (pad->32), plain mvlinear (flat weights)
    k_simplex<<<cdiv((long)N0 * 30 * 8, 256), 256, 0, stream>>>(pos, vel, charges, xi0, Fb, N0, 1, 0);
    {
        GemmSeg f0{Fb, nullptr, FLD, 0, N0 * FLD, 0};
        mv_gemm(stream, 1, 0, f0, z, z, cl0.w, cl0.b, Tb, N0, 30, HIDC);
    }
    k_perm_sum_set<<<cdiv((long)8 * N0 * 32, 256), 256, 0, stream>>>(x, Tb, id0, N0, 1);

    // d1: 24000 rows, 60 ch (pad->64), cemlp (1 block)
    k_simplex<<<cdiv((long)N1 * 2 * 60 * 8, 256), 256, 0, stream>>>(pos, vel, charges, xi1, Fb, N1, 2, 1);
    {
        int rows = N1 * 2, PS = rows * FLD;
        GemmSeg f0{Fb, nullptr, FLD, 0,  PS, 0};
        GemmSeg f1{Fb, nullptr, FLD, 32, PS, 0};
        run_block(stream, cl1b, rows, f0, f1, z, 2, 60, Tb, Rb, Lb);
    }
    k_perm_sum_set<<<cdiv((long)8 * N1 * 32, 256), 256, 0, stream>>>(x, Tb, id1, N1, 2);

    // d2: 36000 rows, 90 ch (pad->96), cemlp (2 blocks)
    k_simplex<<<cdiv((long)N2 * 6 * 90 * 8, 256), 256, 0, stream>>>(pos, vel, charges, xi2, Fb, N2, 6, 2);
    {
        int rows = N2 * 6, PS = rows * FLD;
        GemmSeg f0{Fb, nullptr, FLD, 0,  PS, 0};
        GemmSeg f1{Fb, nullptr, FLD, 32, PS, 0};
        GemmSeg f2{Fb, nullptr, FLD, 64, PS, 0};
        run_block(stream, cl2b[0], rows, f0, f1, f2, 3, 90, Tb, Rb, Lb);
        run_block(stream, cl2b[1], rows, seg32(Tb, nullptr, rows), z, z, 1, 32, Rb, Lb, Fb);
    }
    k_perm_sum_set<<<cdiv((long)8 * N2 * 32, 256), 256, 0, stream>>>(x, Rb, id2, N2, 6);

    // feat_emb: [x | node_attr] (35 ch, flat weights) -> x
    mv_gemm(stream, 2, 0, seg32(x, nullptr, NN), na, z, feat.w, feat.b, Tb, NN, 35, HIDC);
    k_perm_sum_set<<<cdiv((long)8 * NN * 32, 256), 256, 0, stream>>>(x, Tb, nullptr, NN, 1);

    // ---- 5 EGCL layers ----
    for (int l = 0; l < 5; ++l) {
        // edge cemlp on [x[dst] | x[src] | edge_attr] (gathers fused into GEMM A)
        run_block(stream, eg_e[l][0], EE,
                  seg32(x, dst, NN), seg32(x, src, NN), ea, 3, 70, Tb, Rb, Lb);
        run_block(stream, eg_e[l][1], EE, seg32(Tb, nullptr, EE), z, z, 1, 32, Rb, Lb, Fb);
        // segment-sum messages into agg
        k_zero<<<cdiv((long)8 * NN * 32, 256), 256, 0, stream>>>(agg, 8 * NN * 32);
        k_seg_add<<<cdiv((long)8 * EE * 32, 256), 256, 0, stream>>>(Rb, dst, agg, 8 * EE * 32);
        // node cemlp on [x | agg | node_attr]
        run_block(stream, eg_n[l][0], NN,
                  seg32(x, nullptr, NN), seg32(agg, nullptr, NN), na, 3, 67, Tb, Rb, Lb);
        run_block(stream, eg_n[l][1], NN, seg32(Tb, nullptr, NN), z, z, 1, 32, Rb, Lb, Fb);
        k_add<<<cdiv((long)8 * NN * 32, 256), 256, 0, stream>>>(x, Rb, 8 * NN * 32);
    }

    // ---- projection head ----
    run_block(stream, pmlp, N0, seg32(x, id0, NN), z, z, 1, 32, Tb, Rb, Lb);
    mv_gemm(stream, 1, 1, seg32(Tb, nullptr, N0), z, z, plin.w, plin.b, Rb, N0, 32, 10);
    k_out<<<cdiv((long)N0 * 30, 256), 256, 0, stream>>>(loc, id0, Rb, (float*)d_out);
}
